// TemporalEncoderSSM_24833500905966
// MI455X (gfx1250) — matrix operand order repaired and checked by reading, not compiled, not run on hardware
//
#include <hip/hip_runtime.h>
#include <hip/hip_bf16.h>

typedef __attribute__((ext_vector_type(16))) _Float16 v16h;
typedef __attribute__((ext_vector_type(8)))  _Float16 v8h;
typedef __attribute__((ext_vector_type(8)))  float    v8f;

#define SEQS   128
#define LSEQ   512
#define TOKENS 65536      // SEQS * LSEQ
#define DMODEL 128
#define DINNER 256
#define DSTATE 16
#define DTRANK 8
#define DBL_LD 48         // padded 40 -> 48 (3 WMMA n-tiles)
#define DT_KPAD 32        // dt-proj K padded 8 -> 32

// ---------------------------------------------------------------------------
// h[seq, t, d] = x[b, t, v] * enc_w[d] + enc_b[d],  seq = b*32 + v
// ---------------------------------------------------------------------------
__global__ void encode_kernel(const float* __restrict__ x, const float* __restrict__ ew,
                              const float* __restrict__ eb, float* __restrict__ h) {
  int idx = blockIdx.x * blockDim.x + threadIdx.x;          // TOKENS*DMODEL
  int d   = idx & (DMODEL - 1);
  int t   = (idx >> 7) & (LSEQ - 1);
  int seq = idx >> 16;
  int b = seq >> 5, v = seq & 31;
  float xv = x[(b * LSEQ + t) * 32 + v];
  h[idx] = xv * ew[d] + eb[d];
}

// ---------------------------------------------------------------------------
// Weight prep: B (Kreal x Nreal, f32 row-major) -> Bt (Npad x Kpad, f16),
// transposed + zero padded. Makes GEMM B-fragment loads contiguous b128.
// ---------------------------------------------------------------------------
__global__ void wprep_kernel(const float* __restrict__ B, int Kreal, int Kpad,
                             int Nreal, _Float16* __restrict__ Bt) {
  int idx = blockIdx.x * blockDim.x + threadIdx.x;          // Npad*Kpad
  int k = idx % Kpad, n = idx / Kpad;
  Bt[idx] = (k < Kreal && n < Nreal) ? (_Float16)B[k * Nreal + n] : (_Float16)0.0f;
}

// ---------------------------------------------------------------------------
// RMSNorm over rows of 128. One wave32 per row, float4 per lane, shfl reduce.
// ---------------------------------------------------------------------------
__global__ void rmsnorm_kernel(const float* __restrict__ in, const float* __restrict__ w,
                               _Float16* __restrict__ out16, float* __restrict__ out32,
                               int nrows) {
  int gid  = blockIdx.x * blockDim.x + threadIdx.x;
  int row  = gid >> 5;
  int lane = gid & 31;
  if (row >= nrows) return;
  const float* r = in + (size_t)row * DMODEL;
  int base = lane * 4;
  float x0 = r[base], x1 = r[base + 1], x2 = r[base + 2], x3 = r[base + 3];
  float ss = x0 * x0 + x1 * x1 + x2 * x2 + x3 * x3;
  #pragma unroll
  for (int off = 16; off > 0; off >>= 1) ss += __shfl_xor(ss, off, 32);
  float s = rsqrtf(ss * (1.0f / 128.0f) + 1e-5f);
  float y0 = x0 * s * w[base],     y1 = x1 * s * w[base + 1];
  float y2 = x2 * s * w[base + 2], y3 = x3 * s * w[base + 3];
  if (out16) {
    _Float16* o = out16 + (size_t)row * DMODEL + base;
    o[0] = (_Float16)y0; o[1] = (_Float16)y1; o[2] = (_Float16)y2; o[3] = (_Float16)y3;
  }
  if (out32) {
    float* o = out32 + (size_t)row * DMODEL + base;
    o[0] = y0; o[1] = y1; o[2] = y2; o[3] = y3;
  }
}

// ---------------------------------------------------------------------------
// WMMA GEMM, compile-time K (multiple of 32), NT 16-wide N-tiles per wave.
// A: (M x K) f16 row-major (lda == K).  Bt: (N x K) f16 (pre-transposed).
// Fragment loads are contiguous b128 per the gfx1250 WMMA VGPR layouts:
//   A lane(m=lane&15, hi=lane>>4): halves 0-7 = K[kk+8*hi ..], 8-15 = K[kk+16+8*hi ..]
//   B lane(n=lane&15, hi):         halves 0-7 = K[kk+8*hi ..], 8-15 = K[kk+16+8*hi ..] of column n
// Optional residual add and f16/f32 store.
// ---------------------------------------------------------------------------
template <int K, int NT>
__global__ void wmma_gemm(const _Float16* __restrict__ A, const _Float16* __restrict__ Bt,
                          float* __restrict__ C32, _Float16* __restrict__ C16,
                          const float* __restrict__ addbase, int ldc, int ng) {
  int wid  = (blockIdx.x * blockDim.x + threadIdx.x) >> 5;
  int lane = threadIdx.x & 31;
  int tm = wid / ng, tg = wid % ng;
  int m0 = tm * 16, n0 = tg * (16 * NT);
  int lh = lane & 15, hi = lane >> 4;

  const _Float16* arow = A + (size_t)(m0 + lh) * K + hi * 8;
  const _Float16* brows[NT];
  #pragma unroll
  for (int nt = 0; nt < NT; ++nt)
    brows[nt] = Bt + (size_t)(n0 + nt * 16 + lh) * K + hi * 8;

  v8f acc[NT];
  #pragma unroll
  for (int nt = 0; nt < NT; ++nt) acc[nt] = (v8f){0.f,0.f,0.f,0.f,0.f,0.f,0.f,0.f};

  #pragma unroll
  for (int kk = 0; kk < K; kk += 32) {
    v8h alo = *(const v8h*)(arow + kk);
    v8h ahi = *(const v8h*)(arow + kk + 16);
    v16h a = __builtin_shufflevector(alo, ahi, 0,1,2,3,4,5,6,7,8,9,10,11,12,13,14,15);
    #pragma unroll
    for (int nt = 0; nt < NT; ++nt) {
      v8h blo = *(const v8h*)(brows[nt] + kk);
      v8h bhi = *(const v8h*)(brows[nt] + kk + 16);
      v16h b = __builtin_shufflevector(blo, bhi, 0,1,2,3,4,5,6,7,8,9,10,11,12,13,14,15);
      acc[nt] = __builtin_amdgcn_wmma_f32_16x16x32_f16(false, a, false, b, (short)0,
                                                       acc[nt], false, false);
    }
  }

  #pragma unroll
  for (int nt = 0; nt < NT; ++nt) {
    int n = n0 + nt * 16 + lh;
    #pragma unroll
    for (int r = 0; r < 8; ++r) {
      int m = m0 + r + 8 * hi;
      size_t off = (size_t)m * ldc + n;
      float v = acc[nt][r];
      if (addbase) v += addbase[off];
      if (C32) C32[off] = v;
      if (C16) C16[off] = (_Float16)v;
    }
  }
}

// ---------------------------------------------------------------------------
// Depthwise causal conv (4 taps) over xi = xz[:, :256], then SiLU -> u (f16).
// ---------------------------------------------------------------------------
__global__ void conv_silu_kernel(const _Float16* __restrict__ xz, const float* __restrict__ cw,
                                 const float* __restrict__ cb, _Float16* __restrict__ u) {
  int idx = blockIdx.x * blockDim.x + threadIdx.x;          // TOKENS*DINNER
  int d   = idx & (DINNER - 1);
  int tok = idx >> 8;
  int t   = tok & (LSEQ - 1);
  float acc = cb[d];
  #pragma unroll
  for (int k = 0; k < 4; ++k) {
    int tt = t - 3 + k;
    if (tt >= 0) acc += (float)xz[(size_t)(tok - 3 + k) * 512 + d] * cw[d * 4 + k];
  }
  float sg = acc / (1.0f + __expf(-acc));
  u[idx] = (_Float16)sg;
}

// dt cols (first 8 of dbl row, ld 48) -> zero-padded (TOKENS x 32) f16 GEMM A
__global__ void dtcvt_kernel(const float* __restrict__ dbl, _Float16* __restrict__ dtin) {
  int idx = blockIdx.x * blockDim.x + threadIdx.x;          // TOKENS*DT_KPAD
  int k = idx & (DT_KPAD - 1), tok = idx >> 5;
  dtin[idx] = (k < DTRANK) ? (_Float16)dbl[(size_t)tok * DBL_LD + k] : (_Float16)0.0f;
}

// ---------------------------------------------------------------------------
// Selective scan. One block per sequence (256 threads = channels), 16-entry
// state per thread in registers, B/C broadcast via LDS. Fuses softplus(dt),
// y + D*u, and the z-gate: g = (y + D*u) * silu(z), stored f16 for out-proj.
// ---------------------------------------------------------------------------
__global__ void scan_kernel(const float* __restrict__ dbl, const float* __restrict__ dtmat,
                            const _Float16* __restrict__ u, const _Float16* __restrict__ xz,
                            const float* __restrict__ dt_bias, const float* __restrict__ A_log,
                            const float* __restrict__ Dv, _Float16* __restrict__ g) {
  int seq = blockIdx.x;
  int d   = threadIdx.x;
  __shared__ float bc[32];                                   // B[16] then C[16]
  float negA[DSTATE], hs[DSTATE];
  #pragma unroll
  for (int s = 0; s < DSTATE; ++s) {
    negA[s] = -__expf(A_log[d * DSTATE + s]);
    hs[s] = 0.0f;
  }
  float bias = dt_bias[d], Dd = Dv[d];
  for (int t = 0; t < LSEQ; ++t) {
    int tok = seq * LSEQ + t;
    if (d < 32) bc[d] = dbl[(size_t)tok * DBL_LD + DTRANK + d];
    __syncthreads();
    float dtv = dtmat[(size_t)tok * DINNER + d] + bias;
    dtv = (dtv > 0.0f) ? (dtv + log1pf(__expf(-dtv))) : log1pf(__expf(dtv));  // softplus
    float uv = (float)u[(size_t)tok * DINNER + d];
    float du = dtv * uv;
    float y = 0.0f;
    #pragma unroll
    for (int s = 0; s < DSTATE; ++s) {
      float dA = __expf(dtv * negA[s]);
      hs[s] = dA * hs[s] + du * bc[s];
      y += hs[s] * bc[16 + s];
    }
    float zv = (float)xz[(size_t)tok * 512 + DINNER + d];
    float gv = (y + Dd * uv) * (zv / (1.0f + __expf(-zv)));
    g[(size_t)tok * DINNER + d] = (_Float16)gv;
    __syncthreads();
  }
}

// ---------------------------------------------------------------------------
extern "C" void kernel_launch(void* const* d_in, const int* in_sizes, int n_in,
                              void* d_out, int out_size, void* d_ws, size_t ws_size,
                              hipStream_t stream) {
  (void)in_sizes; (void)n_in; (void)out_size; (void)ws_size;
  const float* x    = (const float*)d_in[0];
  const float* ew   = (const float*)d_in[1];
  const float* eb   = (const float*)d_in[2];
  const float* nw   = (const float*)d_in[3];
  const float* inw  = (const float*)d_in[4];
  const float* cw   = (const float*)d_in[5];
  const float* cb   = (const float*)d_in[6];
  const float* xpw  = (const float*)d_in[7];
  const float* dtpw = (const float*)d_in[8];
  const float* dtb  = (const float*)d_in[9];
  const float* alog = (const float*)d_in[10];
  const float* Dv   = (const float*)d_in[11];
  const float* outw = (const float*)d_in[12];
  const float* fnw  = (const float*)d_in[13];

  char* ws = (char*)d_ws;
  size_t off = 0;
  auto carve = [&](size_t bytes) -> void* {
    void* p = ws + off;
    off += (bytes + 255) & ~(size_t)255;
    return p;
  };
  float*    h     = (float*)   carve((size_t)TOKENS * DMODEL * 4);   // residual stream
  _Float16* xn    = (_Float16*)carve((size_t)TOKENS * DMODEL * 2);   // rmsnorm out
  _Float16* xz    = (_Float16*)carve((size_t)TOKENS * 512    * 2);   // in-proj out
  _Float16* u     = (_Float16*)carve((size_t)TOKENS * DINNER * 2);   // conv+silu
  float*    dbl   = (float*)   carve((size_t)TOKENS * DBL_LD * 4);   // x-proj (padded 48)
  _Float16* dtin  = (_Float16*)carve((size_t)TOKENS * DT_KPAD * 2);  // dt GEMM A (K-pad)
  float*    dtmat = (float*)   carve((size_t)TOKENS * DINNER * 4);   // dt-proj out
  _Float16* g     = (_Float16*)carve((size_t)TOKENS * DINNER * 2);   // gated scan out
  _Float16* btin  = (_Float16*)carve((size_t)512 * DMODEL * 2);      // in_w^T  (512x128)
  _Float16* btxp  = (_Float16*)carve((size_t)DBL_LD * DINNER * 2);   // xproj^T (48x256)
  _Float16* btdt  = (_Float16*)carve((size_t)DINNER * DT_KPAD * 2);  // dtproj^T (256x32)
  _Float16* btout = (_Float16*)carve((size_t)DMODEL * DINNER * 2);   // out_w^T (128x256)

  encode_kernel<<<TOKENS * DMODEL / 256, 256, 0, stream>>>(x, ew, eb, h);

  for (int l = 0; l < 2; ++l) {
    // Transpose/pad this layer's weights to f16 (tiny)
    wprep_kernel<<<512 * DMODEL / 256, 256, 0, stream>>>(
        inw + (size_t)l * DMODEL * 512, DMODEL, DMODEL, 512, btin);
    wprep_kernel<<<DBL_LD * DINNER / 256, 256, 0, stream>>>(
        xpw + (size_t)l * DINNER * 40, DINNER, DINNER, 40, btxp);
    wprep_kernel<<<DINNER * DT_KPAD / 256, 256, 0, stream>>>(
        dtpw + (size_t)l * DTRANK * DINNER, DTRANK, DT_KPAD, DINNER, btdt);
    wprep_kernel<<<DMODEL * DINNER / 256, 256, 0, stream>>>(
        outw + (size_t)l * DINNER * DMODEL, DINNER, DINNER, DMODEL, btout);

    rmsnorm_kernel<<<TOKENS / 8, 256, 0, stream>>>(h, nw + l * DMODEL, xn, nullptr, TOKENS);

    // in-proj: (65536x128) @ (128x512) -> xz (f16). 8 groups of 4 n-tiles.
    wmma_gemm<DMODEL, 4><<<(TOKENS / 16) * 8 / 8, 256, 0, stream>>>(
        xn, btin, nullptr, xz, nullptr, 512, 8);

    conv_silu_kernel<<<TOKENS * DINNER / 256, 256, 0, stream>>>(
        xz, cw + l * DINNER * 4, cb + l * DINNER, u);

    // x-proj: (65536x256) @ (256x40->48) -> dbl (f32). 1 group of 3 n-tiles.
    wmma_gemm<DINNER, 3><<<(TOKENS / 16) * 1 / 8 * 8 / 8, 256, 0, stream>>>(
        u, btxp, dbl, nullptr, nullptr, DBL_LD, 1);

    dtcvt_kernel<<<TOKENS * DT_KPAD / 256, 256, 0, stream>>>(dbl, dtin);

    // dt-proj: (65536x32pad) @ (32x256) -> dtmat (f32). 4 groups of 4 n-tiles.
    wmma_gemm<DT_KPAD, 4><<<(TOKENS / 16) * 4 / 8, 256, 0, stream>>>(
        dtin, btdt, dtmat, nullptr, nullptr, DINNER, 4);

    scan_kernel<<<SEQS, DINNER, 0, stream>>>(
        dbl, dtmat, u, xz, dtb + l * DINNER, alog + l * DINNER * DSTATE,
        Dv + l * DINNER, g);

    // out-proj + residual: h = h + (65536x256) @ (256x128). 2 groups of 4 n-tiles.
    wmma_gemm<DINNER, 4><<<(TOKENS / 16) * 2 / 8, 256, 0, stream>>>(
        g, btout, h, nullptr, h, DMODEL, 2);
  }

  rmsnorm_kernel<<<TOKENS / 8, 256, 0, stream>>>(h, fnw, nullptr, (float*)d_out, TOKENS);
}
